// GCN_90890097918492
// MI455X (gfx1250) — compile-verified, matrix operand order checked by reading, not run on hardware
//
#include <hip/hip_runtime.h>
#include <hip/hip_bf16.h>

typedef __attribute__((ext_vector_type(2)))  float    v2f;
typedef __attribute__((ext_vector_type(8)))  float    v8f;
typedef __attribute__((ext_vector_type(16))) _Float16 v16h;

// ---------------------------------------------------------------------------
// init: zero layer-1 accumulator (padded N*16), deg=1 for real nodes (self
// loop) / 0 for padding, zero z, zero output accumulator.
// ---------------------------------------------------------------------------
__global__ void gcn_init_kernel(float* __restrict__ h2acc, float* __restrict__ deg,
                                float* __restrict__ z, float* __restrict__ out,
                                int N, int paddedN) {
    int i = blockIdx.x * blockDim.x + threadIdx.x;
    if (i < paddedN * 16) h2acc[i] = 0.0f;
    if (i < paddedN) { deg[i] = (i < N) ? 1.0f : 0.0f; z[i] = 0.0f; }
    if (i < N) out[i] = 0.0f;
}

// deg[dst] += 1 per edge (self-loops already folded in via init=1)
__global__ void gcn_degree_kernel(const int* __restrict__ dst, float* __restrict__ deg, int E) {
    int e = blockIdx.x * blockDim.x + threadIdx.x;
    if (e < E) atomicAdd(&deg[dst[e]], 1.0f);
}

// dinv = rsqrt(deg) in place (deg >= 1 always for real nodes)
__global__ void gcn_rsqrt_kernel(float* __restrict__ dinv, int N) {
    int n = blockIdx.x * blockDim.x + threadIdx.x;
    if (n < N) dinv[n] = rsqrtf(dinv[n]);
}

// Layer-1 edge scatter: 16 lanes per edge (one feature each).
// h[s,f] = x[s] * W1[f] computed on the fly (x is [N,1]), never materialized.
__global__ void gcn_agg1_kernel(const int* __restrict__ src, const int* __restrict__ dst,
                                const float* __restrict__ dinv, const float* __restrict__ x,
                                const float* __restrict__ W1, float* __restrict__ h2acc, int E) {
    int t = blockIdx.x * blockDim.x + threadIdx.x;
    int e = t >> 4;
    if (e >= E) return;
    int f = t & 15;
    int s = src[e];
    int d = dst[e];
    float w = dinv[s] * dinv[d];
    atomicAdd(&h2acc[(size_t)d * 16 + f], w * x[s] * W1[f]);
}

// Pointwise: add self-loop diagonal term + bias, ReLU, in place.
__global__ void gcn_pointwise1_kernel(float* __restrict__ h2, const float* __restrict__ dinv,
                                      const float* __restrict__ x, const float* __restrict__ W1,
                                      const float* __restrict__ b1, int N) {
    int t = blockIdx.x * blockDim.x + threadIdx.x;
    int n = t >> 4;
    if (n >= N) return;
    int f = t & 15;
    float di = dinv[n];
    float v = h2[(size_t)n * 16 + f] + di * di * x[n] * W1[f] + b1[f];
    h2[(size_t)n * 16 + f] = fmaxf(v, 0.0f);
}

// ---------------------------------------------------------------------------
// Layer-2 GEMV z = h2[N,16] @ W2[16,1] via WMMA, one 16-node tile per wave.
// W2 is broadcast to all 16 B columns, so every column of D equals the tile's
// z vector (full f32 precision via V_WMMA_F32_16X16X4_F32, K=16 in 4 steps).
// A layout (16x4 f32): lanes 0-15 = rows M, VGPR0/1 = K0/K1 (low half) or
// K2/K3 (high half). D layout: VGPR r holds M=r (lanes 0-15) / M=r+8.
// ---------------------------------------------------------------------------
__global__ void gcn_gemv_wmma_kernel(const float* __restrict__ h2, const float* __restrict__ W2,
                                     float* __restrict__ z, int n_tiles) {
    int lane = threadIdx.x & 31;
    int wave = (blockIdx.x * blockDim.x + threadIdx.x) >> 5;
    if (wave >= n_tiles) return;              // wave-uniform: EXEC stays all-1s for WMMA
    int base = wave * 16;
    int m  = lane & 15;
    int kh = lane >> 4;                        // 0 = low lane half, 1 = high
    const float* row = h2 + (size_t)(base + m) * 16;
    v8f acc = {0.f, 0.f, 0.f, 0.f, 0.f, 0.f, 0.f, 0.f};

#if __has_builtin(__builtin_amdgcn_wmma_f32_16x16x4_f32)
#pragma unroll
    for (int kc = 0; kc < 4; ++kc) {
        v2f a, b;
        a.x = row[kc * 4 + kh * 2 + 0];
        a.y = row[kc * 4 + kh * 2 + 1];
        b.x = W2[kc * 4 + kh * 2 + 0];         // every B column = W2 chunk
        b.y = W2[kc * 4 + kh * 2 + 1];
        acc = __builtin_amdgcn_wmma_f32_16x16x4_f32(
            /*neg_a=*/false, a, /*neg_b=*/false, b,
            /*c_mod=*/(short)0, acc, /*reuse_a=*/false, /*reuse_b=*/false);
    }
#else
    // Fallback: confirmed f16 WMMA, K=16 zero-padded to 32.
    // 16-bit A 16x32: low lanes elems0-7 = K0-7, elems8-15 = K16-23 (pad=0);
    // high lanes elems0-7 = K8-15. B 32x16: low lanes elems0-15 = W2[0-15] (K0-7,K16-23), high 0.
    v16h a, b;
#pragma unroll
    for (int j = 0; j < 8; ++j) {
        a[j]     = (_Float16)row[kh * 8 + j];
        a[8 + j] = (_Float16)0.0f;
        b[j]     = (lane < 16) ? (_Float16)W2[j]     : (_Float16)0.0f;
        b[8 + j] = (lane < 16) ? (_Float16)W2[8 + j] : (_Float16)0.0f;
    }
    acc = __builtin_amdgcn_wmma_f32_16x16x32_f16(
        false, a, false, b, (short)0, acc, false, false);
#endif

    // All columns identical -> lane 0 holds z[base+0..7] in acc[0..7],
    // lane 16 holds z[base+8..15].
    if (m == 0) {
        float* zp = z + base + kh * 8;
        zp[0] = acc[0]; zp[1] = acc[1]; zp[2] = acc[2]; zp[3] = acc[3];
        zp[4] = acc[4]; zp[5] = acc[5]; zp[6] = acc[6]; zp[7] = acc[7];
    }
}

// Layer-2 edge scatter: one f32 atomic per edge directly into d_out.
__global__ void gcn_agg2_kernel(const int* __restrict__ src, const int* __restrict__ dst,
                                const float* __restrict__ dinv, const float* __restrict__ z,
                                float* __restrict__ out, int E) {
    int e = blockIdx.x * blockDim.x + threadIdx.x;
    if (e >= E) return;
    int s = src[e];
    int d = dst[e];
    atomicAdd(&out[d], dinv[s] * dinv[d] * z[s]);
}

// Final: self-loop diagonal + bias.
__global__ void gcn_final_kernel(float* __restrict__ out, const float* __restrict__ dinv,
                                 const float* __restrict__ z, const float* __restrict__ b2, int N) {
    int n = blockIdx.x * blockDim.x + threadIdx.x;
    if (n >= N) return;
    float di = dinv[n];
    out[n] += di * di * z[n] + b2[0];
}

extern "C" void kernel_launch(void* const* d_in, const int* in_sizes, int n_in,
                              void* d_out, int out_size, void* d_ws, size_t ws_size,
                              hipStream_t stream) {
    const float* x  = (const float*)d_in[0];
    const float* W1 = (const float*)d_in[1];
    const float* b1 = (const float*)d_in[2];
    const float* W2 = (const float*)d_in[3];
    const float* b2 = (const float*)d_in[4];
    const int*   ei = (const int*)d_in[5];   // edge_index [2, E] int32
    int N = in_sizes[0];                      // x is [N,1]
    int E = in_sizes[5] / 2;
    const int* src = ei;
    const int* dst = ei + E;
    int paddedN = (N + 15) & ~15;

    char* ws = (char*)d_ws;
    float* h2   = (float*)ws; ws += (size_t)paddedN * 16 * sizeof(float); // [paddedN,16]
    float* dinv = (float*)ws; ws += (size_t)paddedN * sizeof(float);      // deg -> dinv
    float* z    = (float*)ws; ws += (size_t)paddedN * sizeof(float);      // [paddedN]
    float* out  = (float*)d_out;

    const int B = 256;
    int tiles = paddedN / 16;

    gcn_init_kernel<<<(paddedN * 16 + B - 1) / B, B, 0, stream>>>(h2, dinv, z, out, N, paddedN);
    gcn_degree_kernel<<<(E + B - 1) / B, B, 0, stream>>>(dst, dinv, E);
    gcn_rsqrt_kernel<<<(N + B - 1) / B, B, 0, stream>>>(dinv, N);
    gcn_agg1_kernel<<<((long long)E * 16 + B - 1) / B, B, 0, stream>>>(src, dst, dinv, x, W1, h2, E);
    gcn_pointwise1_kernel<<<((long long)N * 16 + B - 1) / B, B, 0, stream>>>(h2, dinv, x, W1, b1, N);
    gcn_gemv_wmma_kernel<<<(tiles * 32 + B - 1) / B, B, 0, stream>>>(h2, W2, z, tiles);
    gcn_agg2_kernel<<<(E + B - 1) / B, B, 0, stream>>>(src, dst, dinv, z, out, E);
    gcn_final_kernel<<<(N + B - 1) / B, B, 0, stream>>>(out, dinv, z, b2, N);
}
